// DVAE_11897059410772
// MI455X (gfx1250) — matrix operand order repaired and checked by reading, not compiled, not run on hardware
//
#include <hip/hip_runtime.h>

#define B_  256
#define N_  96
#define HS_ 384
#define NZ_ 56
#define VS_ 480
#define H3_ 1152

typedef __attribute__((ext_vector_type(16))) __bf16 v16bf;
typedef __attribute__((ext_vector_type(8)))  float  v8f;

union ABu { v16bf v; uint4 q[2]; };

__device__ __forceinline__ unsigned short f2bf(float f) {
    union { float f; unsigned int u; } c; c.f = f;
    unsigned int u = c.u + 0x7FFFu + ((c.u >> 16) & 1u);
    return (unsigned short)(u >> 16);
}
__device__ __forceinline__ float bf2f(unsigned short h) {
    union { unsigned int u; float f; } c; c.u = ((unsigned int)h) << 16;
    return c.f;
}
__device__ __forceinline__ float sigm(float x) { return 1.0f / (1.0f + __expf(-x)); }

// Load one 16x32 A tile / 32x16 B tile (bf16) per the wave32 WMMA VGPR layouts.
__device__ __forceinline__ void ld_a(ABu& t, const unsigned short* p, int k0, int half) {
    t.q[0] = *(const uint4*)(p + k0 + 8 * half);
    t.q[1] = *(const uint4*)(p + k0 + 16 + 8 * half);
}
__device__ __forceinline__ void ld_b(ABu& t, const unsigned short* p, int k0, int half) {
    t.q[0] = *(const uint4*)(p + k0 + 8 * half);
    t.q[1] = *(const uint4*)(p + k0 + 16 + 8 * half);
}

// ---------------------------------------------------------------------------
// Weight conversion f32 -> bf16 (strided source, zero-padded rows)
// ---------------------------------------------------------------------------
__global__ void k_convert(const float* __restrict__ src, unsigned short* __restrict__ dst,
                          int rows, int cols, int stride, int total) {
    int i = blockIdx.x * 256 + threadIdx.x;
    if (i >= total) return;
    int r = i / cols, c = i % cols;
    float v = (r < rows) ? src[(size_t)r * stride + c] : 0.0f;
    dst[i] = f2bf(v);
}

// ---------------------------------------------------------------------------
// Step 1: Hsum[b,h] = sum_{n<v, adj[b,v,n]} G[b,n,h]   (G stored bf16)
// One block per batch b, HS_ threads. Adjacency row cached in LDS.
// ---------------------------------------------------------------------------
__global__ void k_hsum(const int* __restrict__ adj, const unsigned short* __restrict__ G,
                       unsigned short* __restrict__ hsum_bf, float* __restrict__ hsum_f, int v) {
    __shared__ int smask[N_];
    int b = blockIdx.x;
    int h = threadIdx.x;
    if (h < N_) smask[h] = adj[((size_t)b * N_ + v) * N_ + h];
    __syncthreads();
    float acc = 0.0f;
    const unsigned short* gb = G + (size_t)b * N_ * HS_ + h;
    for (int n = 0; n < v; ++n)
        if (smask[n]) acc += bf2f(gb[(size_t)n * HS_]);
    hsum_f[(size_t)b * HS_ + h] = acc;
    hsum_bf[(size_t)b * HS_ + h] = f2bf(acc);
}

// ---------------------------------------------------------------------------
// Step 2+3 fused: per wave, one 16(batch)x16(h) tile of ALL THREE gate chunks
// of gh = Hsum @ Whh^T (B rows at j, j+HS, j+2HS; shared A operand, 3 WMMA
// accumulators), then the full GRU nonlinearity in-register.
// Software-pipelined: iteration it+1's tiles are loaded (into the alternate
// buffer set) BEFORE iteration it's WMMAs issue, so loads overlap matrix ops.
// ---------------------------------------------------------------------------
__global__ void __launch_bounds__(256, 1) k_gemm_gru(const unsigned short* __restrict__ A,
                                                     const unsigned short* __restrict__ Whh,
                                                     const float* __restrict__ x,
                                                     const float* __restrict__ w_ih,
                                                     const float* __restrict__ b_ih,
                                                     const float* __restrict__ b_hh,
                                                     const float* __restrict__ hsum_f,
                                                     unsigned short* __restrict__ hv_bf,
                                                     int v) {
    const int TN   = HS_ / 16;                       // 24 h-tiles
    const int KIT  = HS_ / 32;                       // 12 K-iterations
    int w    = blockIdx.x * 8 + (threadIdx.x >> 5);
    int lane = threadIdx.x & 31;
    int half = lane >> 4, l15 = lane & 15;
    int tm = w / TN, tn = w % TN;
    const unsigned short* ap  = A   + (size_t)(tm * 16 + l15) * HS_;
    const unsigned short* bpr = Whh + (size_t)(tn * 16 + l15) * HS_;
    const unsigned short* bpz = bpr + (size_t)HS_ * HS_;     // rows j+HS
    const unsigned short* bpn = bpz + (size_t)HS_ * HS_;     // rows j+2HS

    ABu a[2], br[2], bz[2], bn[2];
    v8f cr = {}, cz = {}, cn = {};

    ld_a(a[0], ap, 0, half);
    ld_b(br[0], bpr, 0, half);
    ld_b(bz[0], bpz, 0, half);
    ld_b(bn[0], bpn, 0, half);
#pragma unroll
    for (int it = 0; it < KIT; ++it) {
        int cur = it & 1, nxt = cur ^ 1;
        if (it + 1 < KIT) {
            int k1 = (it + 1) * 32;
            __builtin_prefetch(bpr + k1 + 96, 0, 1);         // global_prefetch_b8
            ld_a(a[nxt], ap, k1, half);
            ld_b(br[nxt], bpr, k1, half);
            ld_b(bz[nxt], bpz, k1, half);
            ld_b(bn[nxt], bpn, k1, half);
        }
        cr = __builtin_amdgcn_wmma_f32_16x16x32_bf16(false, a[cur].v, false, br[cur].v,
                                                     (short)0, cr, false, false);
        cz = __builtin_amdgcn_wmma_f32_16x16x32_bf16(false, a[cur].v, false, bz[cur].v,
                                                     (short)0, cz, false, false);
        cn = __builtin_amdgcn_wmma_f32_16x16x32_bf16(false, a[cur].v, false, bn[cur].v,
                                                     (short)0, cn, false, false);
    }
    // GRU epilogue. D layout: element r -> batch b = m0 + r, column h = j.
    int j  = tn * 16 + l15;
    int m0 = tm * 16 + 8 * half;
    float wir = w_ih[j], wiz = w_ih[j + HS_], win = w_ih[j + 2 * HS_];
    float bir = b_ih[j], biz = b_ih[j + HS_], bin = b_ih[j + 2 * HS_];
    float bhr = b_hh[j], bhz = b_hh[j + HS_], bhn = b_hh[j + 2 * HS_];
#pragma unroll
    for (int r = 0; r < 8; ++r) {
        int   b  = m0 + r;
        float xv = x[(size_t)b * N_ + v];
        float rr = sigm(xv * wir + bir + cr[r] + bhr);
        float zz = sigm(xv * wiz + biz + cz[r] + bhz);
        float nn = tanhf(xv * win + bin + rr * (cn[r] + bhn));
        float hs = hsum_f[(size_t)b * HS_ + j];
        float hv = (1.0f - zz) * nn + zz * hs;
        hv_bf[(size_t)b * HS_ + j] = f2bf(hv);
    }
}

// ---------------------------------------------------------------------------
// Step 4: G[:,v,:] = sigmoid(hv@WgH^T + bg + Wg[:,HS+v]) * (hv@WmH^T + Wm[:,HS+v])
// Each wave: 16(batch) x 32(j) for BOTH matrices -> 4 accumulators sharing A.
// Software-pipelined double-buffer as above.
// ---------------------------------------------------------------------------
__global__ void __launch_bounds__(256, 1) k_gate(const unsigned short* __restrict__ hvb,
                                                 const unsigned short* __restrict__ WgHb,
                                                 const unsigned short* __restrict__ WmHb,
                                                 const float* __restrict__ Wg,
                                                 const float* __restrict__ Wm,
                                                 const float* __restrict__ bg,
                                                 unsigned short* __restrict__ G, int v) {
    const int TN  = HS_ / 32;                        // 12 j-tiles of width 32
    const int KIT = HS_ / 32;                        // 12 K-iterations
    int w    = blockIdx.x * 8 + (threadIdx.x >> 5);
    int lane = threadIdx.x & 31;
    int half = lane >> 4, l15 = lane & 15;
    int tm = w / TN, tn = w % TN;
    int j0 = tn * 32;
    const unsigned short* ap  = hvb  + (size_t)(tm * 16 + l15) * HS_;
    const unsigned short* gp0 = WgHb + (size_t)(j0 + l15) * HS_;
    const unsigned short* gp1 = WgHb + (size_t)(j0 + 16 + l15) * HS_;
    const unsigned short* mp0 = WmHb + (size_t)(j0 + l15) * HS_;
    const unsigned short* mp1 = WmHb + (size_t)(j0 + 16 + l15) * HS_;

    ABu a[2], tg0[2], tg1[2], tm0[2], tm1[2];
    v8f cg0 = {}, cg1 = {}, cm0 = {}, cm1 = {};

    ld_a(a[0], ap, 0, half);
    ld_b(tg0[0], gp0, 0, half);
    ld_b(tg1[0], gp1, 0, half);
    ld_b(tm0[0], mp0, 0, half);
    ld_b(tm1[0], mp1, 0, half);
#pragma unroll
    for (int it = 0; it < KIT; ++it) {
        int cur = it & 1, nxt = cur ^ 1;
        if (it + 1 < KIT) {
            int k1 = (it + 1) * 32;
            __builtin_prefetch(gp0 + k1 + 96, 0, 1);
            ld_a(a[nxt], ap, k1, half);
            ld_b(tg0[nxt], gp0, k1, half);
            ld_b(tg1[nxt], gp1, k1, half);
            ld_b(tm0[nxt], mp0, k1, half);
            ld_b(tm1[nxt], mp1, k1, half);
        }
        cg0 = __builtin_amdgcn_wmma_f32_16x16x32_bf16(false, a[cur].v, false, tg0[cur].v,
                                                      (short)0, cg0, false, false);
        cg1 = __builtin_amdgcn_wmma_f32_16x16x32_bf16(false, a[cur].v, false, tg1[cur].v,
                                                      (short)0, cg1, false, false);
        cm0 = __builtin_amdgcn_wmma_f32_16x16x32_bf16(false, a[cur].v, false, tm0[cur].v,
                                                      (short)0, cm0, false, false);
        cm1 = __builtin_amdgcn_wmma_f32_16x16x32_bf16(false, a[cur].v, false, tm1[cur].v,
                                                      (short)0, cm1, false, false);
    }
    int m0 = tm * 16 + 8 * half;
#pragma unroll
    for (int jj = 0; jj < 2; ++jj) {
        int j = j0 + jj * 16 + l15;
        const v8f& cg = jj ? cg1 : cg0;
        const v8f& cm = jj ? cm1 : cm0;
        float bgj = bg[j] + Wg[(size_t)j * VS_ + HS_ + v];   // one-hot hcat column
        float bmj =         Wm[(size_t)j * VS_ + HS_ + v];
#pragma unroll
        for (int r = 0; r < 8; ++r) {
            float g = sigm(cg[r] + bgj) * (cm[r] + bmj);
            G[((size_t)(m0 + r) * N_ + v) * HS_ + j] = f2bf(g);
        }
    }
}

// ---------------------------------------------------------------------------
// Final head: mu = Hg(256 x 384, bf16) @ Wfc^T (56 x 384, zero-padded to 64) + bfc
// ---------------------------------------------------------------------------
__global__ void __launch_bounds__(256, 1) k_final(const unsigned short* __restrict__ Hgb,
                                                  const unsigned short* __restrict__ Wfcb,
                                                  const float* __restrict__ bfc,
                                                  float* __restrict__ out) {
    const int TN  = 64 / 16;                         // 4 tiles (NZ padded 56 -> 64)
    const int KIT = HS_ / 32;
    int w    = blockIdx.x * 8 + (threadIdx.x >> 5);
    int lane = threadIdx.x & 31;
    int half = lane >> 4, l15 = lane & 15;
    int tm = w / TN, tn = w % TN;
    const unsigned short* ap = Hgb  + (size_t)(tm * 16 + l15) * HS_;
    const unsigned short* bp = Wfcb + (size_t)(tn * 16 + l15) * HS_;

    ABu a[2], b[2];
    v8f acc = {};
    ld_a(a[0], ap, 0, half);
    ld_b(b[0], bp, 0, half);
#pragma unroll
    for (int it = 0; it < KIT; ++it) {
        int cur = it & 1, nxt = cur ^ 1;
        if (it + 1 < KIT) {
            int k1 = (it + 1) * 32;
            ld_a(a[nxt], ap, k1, half);
            ld_b(b[nxt], bp, k1, half);
        }
        acc = __builtin_amdgcn_wmma_f32_16x16x32_bf16(false, a[cur].v, false, b[cur].v,
                                                      (short)0, acc, false, false);
    }
    int j  = tn * 16 + l15;
    int m0 = tm * 16 + 8 * half;
    if (j < NZ_) {
        float bj = bfc[j];
#pragma unroll
        for (int r = 0; r < 8; ++r)
            out[(size_t)(m0 + r) * NZ_ + j] = acc[r] + bj;
    }
}

// ---------------------------------------------------------------------------
// Workspace layout (all L2-resident: ~21 MB total)
// ---------------------------------------------------------------------------
static constexpr size_t al256(size_t x) { return (x + 255) & ~(size_t)255; }
static constexpr size_t OFF_G    = 0;                                        // bf16 B*N*HS
static constexpr size_t OFF_HSB  = al256(OFF_G    + (size_t)B_ * N_ * HS_ * 2);
static constexpr size_t OFF_HSF  = al256(OFF_HSB  + (size_t)B_ * HS_ * 2);
static constexpr size_t OFF_HVB  = al256(OFF_HSF  + (size_t)B_ * HS_ * 4);
static constexpr size_t OFF_WHH  = al256(OFF_HVB  + (size_t)B_ * HS_ * 2);
static constexpr size_t OFF_WGH  = al256(OFF_WHH  + (size_t)H3_ * HS_ * 2);
static constexpr size_t OFF_WMH  = al256(OFF_WGH  + (size_t)HS_ * HS_ * 2);
static constexpr size_t OFF_WFC  = al256(OFF_WMH  + (size_t)HS_ * HS_ * 2);

extern "C" void kernel_launch(void* const* d_in, const int* in_sizes, int n_in,
                              void* d_out, int out_size, void* d_ws, size_t ws_size,
                              hipStream_t stream) {
    const float* x    = (const float*)d_in[0];
    const int*   adj  = (const int*)  d_in[1];
    const float* Wg   = (const float*)d_in[2];
    const float* bg   = (const float*)d_in[3];
    const float* Wm   = (const float*)d_in[4];
    const float* w_ih = (const float*)d_in[5];
    const float* w_hh = (const float*)d_in[6];
    const float* b_ih = (const float*)d_in[7];
    const float* b_hh = (const float*)d_in[8];
    const float* Wfc  = (const float*)d_in[9];
    const float* bfc  = (const float*)d_in[10];
    float* out = (float*)d_out;

    char* ws = (char*)d_ws;
    unsigned short* Gb     = (unsigned short*)(ws + OFF_G);
    unsigned short* hsum_b = (unsigned short*)(ws + OFF_HSB);
    float*          hsum_f = (float*)         (ws + OFF_HSF);
    unsigned short* hv_b   = (unsigned short*)(ws + OFF_HVB);
    unsigned short* Whhb   = (unsigned short*)(ws + OFF_WHH);
    unsigned short* WgHb   = (unsigned short*)(ws + OFF_WGH);
    unsigned short* WmHb   = (unsigned short*)(ws + OFF_WMH);
    unsigned short* Wfcb   = (unsigned short*)(ws + OFF_WFC);

    // One-time (per call) weight conversion to bf16.
    k_convert<<<(H3_ * HS_ + 255) / 256, 256, 0, stream>>>(w_hh, Whhb, H3_, HS_, HS_, H3_ * HS_);
    k_convert<<<(HS_ * HS_ + 255) / 256, 256, 0, stream>>>(Wg,   WgHb, HS_, HS_, VS_, HS_ * HS_);
    k_convert<<<(HS_ * HS_ + 255) / 256, 256, 0, stream>>>(Wm,   WmHb, HS_, HS_, VS_, HS_ * HS_);
    k_convert<<<(64  * HS_ + 255) / 256, 256, 0, stream>>>(Wfc,  Wfcb, NZ_, HS_, HS_, 64 * HS_);

    // Sequential scan over nodes. G rows n are written at step n and only
    // consumed at steps v>n, so no zero-init of G is required.
    for (int v = 0; v < N_; ++v) {
        k_hsum<<<B_, HS_, 0, stream>>>(adj, Gb, hsum_b, hsum_f, v);
        k_gemm_gru<<<(16 * (HS_ / 16)) / 8, 256, 0, stream>>>(hsum_b, Whhb, x, w_ih, b_ih,
                                                              b_hh, hsum_f, hv_b, v);   // 48 blocks
        k_gate<<<(16 * (HS_ / 32)) / 8, 256, 0, stream>>>(hv_b, WgHb, WmHb, Wg, Wm,
                                                          bg, Gb, v);                   // 24 blocks
    }
    // hv_b now holds Hg = H[:, N-1] in bf16.
    k_final<<<(16 * 4) / 8, 256, 0, stream>>>(hv_b, Wfcb, bfc, out);                    // 8 blocks
}
